// PointNet_Plus_seg_33706903339272
// MI455X (gfx1250) — compile-verified, matrix operand order checked by reading, not run on hardware
//
#include <hip/hip_runtime.h>
#include <hip/hip_bf16.h>

// ---------------------------------------------------------------------------
// PointNet++ segmentation, MI455X (gfx1250).
// GEMMs: v_wmma_f32_16x16x32_f16 (wave32 WMMA), f32 accumulate.
// r1/r2 stages fused per-KNN-group in LDS; stage-1 gather uses
// global_load_async_to_lds_b32 (+ s_wait_asynccnt); pooling via __shfl_xor;
// KNN via wave32 argmin rounds. All biases pre-padded so epilogues are
// branch-free (merged b128 bias loads).
// ---------------------------------------------------------------------------

typedef __attribute__((ext_vector_type(16))) _Float16 v16h;
typedef __attribute__((ext_vector_type(8)))  float    v8f;
typedef _Float16 half_t;

#define BATCH 16
#define NPTS  1024
#define N1    512
#define N2    128
#define KNN   64
#define ODIM  21

// ---------------- WMMA helpers -------------------------------------------

__device__ __forceinline__ v8f wmma_step(v16h a, v16h b, v8f c) {
  return __builtin_amdgcn_wmma_f32_16x16x32_f16(false, a, false, b, (short)0, c,
                                                false, false);
}

// Load one 16x32 f16 fragment from a row-major [rows][ld] matrix at (row0,k0).
__device__ __forceinline__ v16h load_frag16(const half_t* __restrict__ base,
                                            int ld, int row0, int k0) {
  int lane = (int)(threadIdx.x & 31);
  const half_t* p =
      base + (size_t)(row0 + (lane & 15)) * ld + k0 + ((lane >> 4) << 3);
  v16h f;
#pragma unroll
  for (int j = 0; j < 8; ++j) {
    int k = 2 * j + ((j >= 4) ? 8 : 0);
    f[2 * j]     = p[k];
    f[2 * j + 1] = p[k + 1];
  }
  return f;
}

// C/D layout: lane holds column p = p0 + (lane&15), rows m = m0+8*(lane>>4)+r.
// bias must be padded to Opad (no guards -> merged vector loads).
__device__ __forceinline__ void store_tile_f16(half_t* __restrict__ Y, int ld,
                                               int m0, int p0, v8f acc,
                                               const float* __restrict__ bias,
                                               int relu) {
  int lane = (int)(threadIdx.x & 31);
  int p  = p0 + (lane & 15);
  int mb = m0 + ((lane >> 4) << 3);
  const float* bp = bias + mb;
#pragma unroll
  for (int r = 0; r < 8; ++r) {
    float v = acc[r] + bp[r];
    if (relu) v = fmaxf(v, 0.f);
    Y[(size_t)p * ld + m0 + ((lane >> 4) << 3) + r] = (half_t)v;
  }
}

// ---------------- wave32 argmin (index tie-break like jax top_k) ----------

__device__ __forceinline__ void wave_argmin(float& d, int& q) {
#pragma unroll
  for (int m = 16; m >= 1; m >>= 1) {
    float od = __shfl_xor(d, m, 32);
    int   oq = __shfl_xor(q, m, 32);
    if (od < d || (od == d && oq < q)) { d = od; q = oq; }
  }
}

// ---------------- weight/bias pack f32 -> padded --------------------------

__global__ void pack_w_kernel(const float* __restrict__ src,
                              half_t* __restrict__ dst, int O, int C, int Opad,
                              int Kpad) {
  int i = blockIdx.x * 256 + threadIdx.x;
  if (i >= Opad * Kpad) return;
  int k = i % Kpad, o = i / Kpad;
  dst[i] = (o < O && k < C) ? (half_t)src[(size_t)o * C + k] : (half_t)0.f;
}

__global__ void pack_b_kernel(const float* __restrict__ src,
                              float* __restrict__ dst, int O, int Opad) {
  int i = blockIdx.x * 256 + threadIdx.x;
  if (i < Opad) dst[i] = (i < O) ? src[i] : 0.f;
}

// ---------------- KNN stage 1 (1024 pts -> 512 centers, r^2 = 4.0) --------

__global__ __launch_bounds__(32) void knn1_kernel(const float* __restrict__ pts,
                                                  int* __restrict__ idx1,
                                                  float* __restrict__ o1) {
  int g = blockIdx.x, b = g >> 9, c = g & 511;
  int lane = (int)threadIdx.x;
  const float* px = pts + (size_t)b * 6 * NPTS;
  float cx = px[c], cy = px[NPTS + c], cz = px[2 * NPTS + c];
  float dl[32];
#pragma unroll
  for (int t = 0; t < 32; ++t) {
    int q = t * 32 + lane;
    float dx = px[q] - cx, dy = px[NPTS + q] - cy, dz = px[2 * NPTS + q] - cz;
    dl[t] = dx * dx + dy * dy + dz * dz;
  }
  int selLo = c, selHi = c;
  for (int r = 0; r < 64; ++r) {
    float bd = 1e30f; int bq = 1 << 30;
#pragma unroll
    for (int t = 0; t < 32; ++t) {
      int q = t * 32 + lane;
      if (dl[t] < bd || (dl[t] == bd && q < bq)) { bd = dl[t]; bq = q; }
    }
    wave_argmin(bd, bq);
    int tmin = bq >> 5, lmin = bq & 31;
#pragma unroll
    for (int t = 0; t < 32; ++t)
      if (lane == lmin && t == tmin) dl[t] = 1e30f;
    int sel = (bd > 4.0f) ? c : bq;
    if (r == lane) selLo = sel;
    if (r - 32 == lane) selHi = sel;
  }
  int* ib = idx1 + (size_t)(b * N1 + c) * KNN;
  ib[lane] = selLo;
  ib[32 + lane] = selHi;
  if (lane < 3)
    o1[((size_t)b * 131 + lane) * N1 + c] = (lane == 0 ? cx : (lane == 1 ? cy : cz));
}

// ---------------- fused stage 1 MLP: 6->64->64->128, max over 64 ----------

__global__ __launch_bounds__(128) void stage1_kernel(
    const float* __restrict__ pts, const int* __restrict__ idx1,
    const half_t* __restrict__ W1, const float* __restrict__ B1,
    const half_t* __restrict__ W2, const float* __restrict__ B2,
    const half_t* __restrict__ W3, const float* __restrict__ B3,
    float* __restrict__ o1) {
  __shared__ float  Gs[64 * 6];   // async-gather staging (f32)
  __shared__ half_t Xs[64 * 32];
  __shared__ half_t Hs[64 * 64];
  __shared__ half_t H2[64 * 64];
  int g = blockIdx.x, b = g >> 9, c = g & 511;
  int tid = (int)threadIdx.x, lane = tid & 31, wv = tid >> 5;
  const float* px = pts + (size_t)b * 6 * NPTS;
  float cx = px[c], cy = px[NPTS + c], cz = px[2 * NPTS + c];
  const int* ib = idx1 + (size_t)(b * N1 + c) * KNN;
  // Async gather: per-lane global -> LDS DMA (ASYNCcnt), then convert.
  for (int e = tid; e < 64 * 6; e += 128) {
    int j = e / 6, k = e % 6;
    int q = ib[j];
    const float* gp = px + (size_t)k * NPTS + q;
    unsigned lds = (unsigned)(unsigned long long)&Gs[e];
    asm volatile("global_load_async_to_lds_b32 %0, %1, off"
                 :: "v"(lds), "v"(gp) : "memory");
  }
  asm volatile("s_wait_asynccnt 0x0" ::: "memory");
  __syncthreads();
  for (int e = tid; e < 64 * 32; e += 128) {
    int j = e >> 5, k = e & 31;
    float v = 0.f;
    if (k < 6) {
      v = Gs[j * 6 + k];
      if (k == 0) v -= cx; else if (k == 1) v -= cy; else if (k == 2) v -= cz;
    }
    Xs[e] = (half_t)v;
  }
  __syncthreads();
  { // L1: 64x64, K=32
    int m0 = wv * 16;
    v16h a = load_frag16(W1, 32, m0, 0);
    for (int nt = 0; nt < 4; ++nt) {
      v16h x = load_frag16(Xs, 32, nt * 16, 0);
      v8f acc = {};
      acc = wmma_step(a, x, acc);
      store_tile_f16(Hs, 64, m0, nt * 16, acc, B1, 1);
    }
  }
  __syncthreads();
  { // L2: 64x64, K=64
    int m0 = wv * 16;
    v16h a0 = load_frag16(W2, 64, m0, 0);
    v16h a1 = load_frag16(W2, 64, m0, 32);
    for (int nt = 0; nt < 4; ++nt) {
      v8f acc = {};
      acc = wmma_step(a0, load_frag16(Hs, 64, nt * 16, 0), acc);
      acc = wmma_step(a1, load_frag16(Hs, 64, nt * 16, 32), acc);
      store_tile_f16(H2, 64, m0, nt * 16, acc, B2, 1);
    }
  }
  __syncthreads();
  // L3: 128x64, K=64, fused ReLU + max-pool over the 64 points
  for (int mi = 0; mi < 2; ++mi) {
    int m0 = (wv * 2 + mi) * 16;
    v16h a0 = load_frag16(W3, 64, m0, 0);
    v16h a1 = load_frag16(W3, 64, m0, 32);
    int mb = m0 + ((lane >> 4) << 3);
    float bb[8];
#pragma unroll
    for (int r = 0; r < 8; ++r) bb[r] = B3[mb + r];
    float vmax[8];
#pragma unroll
    for (int r = 0; r < 8; ++r) vmax[r] = -1e30f;
    for (int nt = 0; nt < 4; ++nt) {
      v8f acc = {};
      acc = wmma_step(a0, load_frag16(H2, 64, nt * 16, 0), acc);
      acc = wmma_step(a1, load_frag16(H2, 64, nt * 16, 32), acc);
#pragma unroll
      for (int r = 0; r < 8; ++r)
        vmax[r] = fmaxf(vmax[r], fmaxf(acc[r] + bb[r], 0.f));
    }
#pragma unroll
    for (int mm = 8; mm >= 1; mm >>= 1)
#pragma unroll
      for (int r = 0; r < 8; ++r)
        vmax[r] = fmaxf(vmax[r], __shfl_xor(vmax[r], mm, 32));
    if ((lane & 15) == 0)
#pragma unroll
      for (int r = 0; r < 8; ++r)
        o1[((size_t)b * 131 + 3 + mb + r) * N1 + c] = vmax[r];
  }
}

// ---------------- KNN stage 2 (512 pts -> 128 centers, r^2 = 8.0) --------

__global__ __launch_bounds__(32) void knn2_kernel(const float* __restrict__ o1,
                                                  int* __restrict__ idx2,
                                                  float* __restrict__ o2f,
                                                  half_t* __restrict__ o2h) {
  int g = blockIdx.x, b = g >> 7, c = g & 127;
  int lane = (int)threadIdx.x;
  const float* ob = o1 + (size_t)b * 131 * N1;
  float cx = ob[c], cy = ob[N1 + c], cz = ob[2 * N1 + c];
  float dl[16];
#pragma unroll
  for (int t = 0; t < 16; ++t) {
    int q = t * 32 + lane;
    float dx = ob[q] - cx, dy = ob[N1 + q] - cy, dz = ob[2 * N1 + q] - cz;
    dl[t] = dx * dx + dy * dy + dz * dz;
  }
  int selLo = c, selHi = c;
  for (int r = 0; r < 64; ++r) {
    float bd = 1e30f; int bq = 1 << 30;
#pragma unroll
    for (int t = 0; t < 16; ++t) {
      int q = t * 32 + lane;
      if (dl[t] < bd || (dl[t] == bd && q < bq)) { bd = dl[t]; bq = q; }
    }
    wave_argmin(bd, bq);
    int tmin = bq >> 5, lmin = bq & 31;
#pragma unroll
    for (int t = 0; t < 16; ++t)
      if (lane == lmin && t == tmin) dl[t] = 1e30f;
    int sel = (bd > 8.0f) ? c : bq;
    if (r == lane) selLo = sel;
    if (r - 32 == lane) selHi = sel;
  }
  int* ib = idx2 + (size_t)(b * N2 + c) * KNN;
  ib[lane] = selLo;
  ib[32 + lane] = selHi;
  if (lane < 3) {
    float v = (lane == 0 ? cx : (lane == 1 ? cy : cz));
    o2f[((size_t)b * 259 + lane) * N2 + c] = v;
    o2h[((size_t)b * N2 + c) * 288 + lane] = (half_t)v;
  }
  for (int ch = 259 + lane; ch < 288; ch += 32)
    o2h[((size_t)b * N2 + c) * 288 + ch] = (half_t)0.f;
}

// ---------------- fused stage 2 MLP: 131->128->128->256, max over 64 ------

__global__ __launch_bounds__(256) void stage2_kernel(
    const float* __restrict__ o1, const int* __restrict__ idx2,
    const half_t* __restrict__ W1, const float* __restrict__ B1,
    const half_t* __restrict__ W2, const float* __restrict__ B2,
    const half_t* __restrict__ W3, const float* __restrict__ B3,
    float* __restrict__ o2f, half_t* __restrict__ o2h) {
  __shared__ half_t Xs[64 * 160];
  __shared__ half_t Hs[64 * 128];
  __shared__ half_t H2[64 * 128];
  int g = blockIdx.x, b = g >> 7, c = g & 127;
  int tid = (int)threadIdx.x, lane = tid & 31, wv = tid >> 5;
  const float* ob = o1 + (size_t)b * 131 * N1;
  float cx = ob[c], cy = ob[N1 + c], cz = ob[2 * N1 + c];
  const int* ib = idx2 + (size_t)(b * N2 + c) * KNN;
  for (int e = tid; e < 64 * 160; e += 256) {
    int j = e / 160, k = e % 160;
    float v = 0.f;
    if (k < 131) {
      int q = ib[j];
      v = ob[(size_t)k * N1 + q];
      if (k == 0) v -= cx; else if (k == 1) v -= cy; else if (k == 2) v -= cz;
    }
    Xs[e] = (half_t)v;
  }
  __syncthreads();
  { // L1: 128 out, K=160 (A frags hoisted: global weight reads once)
    int m0 = wv * 16;
    v16h a[5];
#pragma unroll
    for (int kk = 0; kk < 5; ++kk) a[kk] = load_frag16(W1, 160, m0, kk * 32);
    for (int nt = 0; nt < 4; ++nt) {
      v8f acc = {};
#pragma unroll
      for (int kk = 0; kk < 5; ++kk)
        acc = wmma_step(a[kk], load_frag16(Xs, 160, nt * 16, kk * 32), acc);
      store_tile_f16(Hs, 128, m0, nt * 16, acc, B1, 1);
    }
  }
  __syncthreads();
  { // L2: 128 out, K=128
    int m0 = wv * 16;
    v16h a[4];
#pragma unroll
    for (int kk = 0; kk < 4; ++kk) a[kk] = load_frag16(W2, 128, m0, kk * 32);
    for (int nt = 0; nt < 4; ++nt) {
      v8f acc = {};
#pragma unroll
      for (int kk = 0; kk < 4; ++kk)
        acc = wmma_step(a[kk], load_frag16(Hs, 128, nt * 16, kk * 32), acc);
      store_tile_f16(H2, 128, m0, nt * 16, acc, B2, 1);
    }
  }
  __syncthreads();
  // L3: 256 out, K=128, fused ReLU + max-pool
  for (int mi = 0; mi < 2; ++mi) {
    int m0 = (wv * 2 + mi) * 16;
    v16h a[4];
#pragma unroll
    for (int kk = 0; kk < 4; ++kk) a[kk] = load_frag16(W3, 128, m0, kk * 32);
    int mb = m0 + ((lane >> 4) << 3);
    float bb[8];
#pragma unroll
    for (int r = 0; r < 8; ++r) bb[r] = B3[mb + r];
    float vmax[8];
#pragma unroll
    for (int r = 0; r < 8; ++r) vmax[r] = -1e30f;
    for (int nt = 0; nt < 4; ++nt) {
      v8f acc = {};
#pragma unroll
      for (int kk = 0; kk < 4; ++kk)
        acc = wmma_step(a[kk], load_frag16(H2, 128, nt * 16, kk * 32), acc);
#pragma unroll
      for (int r = 0; r < 8; ++r)
        vmax[r] = fmaxf(vmax[r], fmaxf(acc[r] + bb[r], 0.f));
    }
#pragma unroll
    for (int mm = 8; mm >= 1; mm >>= 1)
#pragma unroll
      for (int r = 0; r < 8; ++r)
        vmax[r] = fmaxf(vmax[r], __shfl_xor(vmax[r], mm, 32));
    if ((lane & 15) == 0)
#pragma unroll
      for (int r = 0; r < 8; ++r) {
        int m = mb + r;
        o2f[((size_t)b * 259 + 3 + m) * N2 + c] = vmax[r];
        o2h[((size_t)b * N2 + c) * 288 + 3 + m] = (half_t)vmax[r];
      }
  }
}

// ---------------- generic WMMA GEMM: Y = act(W X + b) ---------------------
// 16(M) x 64(N) per wave: one A frag reused across 4 N tiles.
// W: [Opad][Kpad] f16.  bias: [Opad] f32 (padded).  X: [B][P][Kpad] f16.

__global__ __launch_bounds__(32) void gemm_wmma_kernel(
    const half_t* __restrict__ W, const float* __restrict__ bias,
    const half_t* __restrict__ X, half_t* __restrict__ Yh,
    float* __restrict__ Yf, int O, int Opad, int Kpad, int P, int relu,
    int cmajor) {
  int b = blockIdx.z, m0 = blockIdx.y * 16, p0 = blockIdx.x * 64;
  const half_t* Xb = X + (size_t)b * P * Kpad;
  v8f acc[4] = {};
  for (int k0 = 0; k0 < Kpad; k0 += 32) {
    if (k0 + 32 < Kpad)  // prefetch next weight chunk (global_prefetch_b8)
      __builtin_prefetch((const void*)(W + (size_t)m0 * Kpad + k0 + 32), 0, 1);
    v16h a = load_frag16(W, Kpad, m0, k0);
#pragma unroll
    for (int t = 0; t < 4; ++t)
      acc[t] = wmma_step(a, load_frag16(Xb, Kpad, p0 + t * 16, k0), acc[t]);
  }
  int lane = (int)(threadIdx.x & 31);
  int mb = m0 + ((lane >> 4) << 3);
  float bb[8];
#pragma unroll
  for (int r = 0; r < 8; ++r) bb[r] = bias[mb + r];
#pragma unroll
  for (int t = 0; t < 4; ++t) {
    int p = p0 + t * 16 + (lane & 15);
#pragma unroll
    for (int r = 0; r < 8; ++r) {
      int m = mb + r;
      float v = acc[t][r] + bb[r];
      if (relu) v = fmaxf(v, 0.f);
      if (Yh) Yh[((size_t)b * P + p) * Opad + m] = (half_t)v;
      if (Yf) {
        if (cmajor) {
          if (m < O) Yf[((size_t)b * O + m) * P + p] = v;
        } else {
          Yf[((size_t)b * P + p) * Opad + m] = v;
        }
      }
    }
  }
}

// ---------------- global max over points (o3) -----------------------------

__global__ void rowmax_kernel(const float* __restrict__ X /*[B][P][C]*/,
                              float* __restrict__ out /*[B][C]*/, int P,
                              int C) {
  int i = blockIdx.x * 256 + threadIdx.x;
  if (i >= BATCH * C) return;
  int b = i / C, ch = i % C;
  const float* xp = X + (size_t)b * P * C + ch;
  float mx = -1e30f;
  for (int p = 0; p < P; ++p) mx = fmaxf(mx, xp[(size_t)p * C]);
  out[(size_t)b * C + ch] = mx;
}

// ---------------- build fp3 input: concat(o2, broadcast o3) ---------------

__global__ void build_fp3in_kernel(const float* __restrict__ o2f,
                                   const float* __restrict__ o3,
                                   half_t* __restrict__ out) {
  size_t i = (size_t)blockIdx.x * 256 + threadIdx.x;
  if (i >= (size_t)BATCH * N2 * 1312) return;
  int ch = (int)(i % 1312);
  size_t pc = i / 1312;
  int p = (int)(pc % N2), b = (int)(pc / N2);
  float v = 0.f;
  if (ch < 259)       v = o2f[((size_t)b * 259 + ch) * N2 + p];
  else if (ch < 1283) v = o3[(size_t)b * 1024 + (ch - 259)];
  out[i] = (half_t)v;
}

// ---------------- 3-NN propagation (coarse 128 -> fine 512) ---------------

__global__ __launch_bounds__(32) void prop2_kernel(
    const float* __restrict__ o1, const float* __restrict__ fpr3 /*[B][128][256]*/,
    half_t* __restrict__ out /*[B][512][416]*/) {
  int g = blockIdx.x, b = g >> 9, p = g & 511;
  int lane = (int)threadIdx.x;
  const float* ob = o1 + (size_t)b * 131 * N1;
  float fx = ob[p], fy = ob[N1 + p], fz = ob[2 * N1 + p];
  float dl[4];
#pragma unroll
  for (int t = 0; t < 4; ++t) {
    int q = t * 32 + lane;
    float dx = ob[q] - fx, dy = ob[N1 + q] - fy, dz = ob[2 * N1 + q] - fz;
    dl[t] = dx * dx + dy * dy + dz * dz;
  }
  float wk[3]; int ck[3];
#pragma unroll
  for (int r = 0; r < 3; ++r) {
    float bd = 1e30f; int bq = 1 << 30;
#pragma unroll
    for (int t = 0; t < 4; ++t) {
      int q = t * 32 + lane;
      if (dl[t] < bd || (dl[t] == bd && q < bq)) { bd = dl[t]; bq = q; }
    }
    wave_argmin(bd, bq);
    int tmin = bq >> 5, lmin = bq & 31;
#pragma unroll
    for (int t = 0; t < 4; ++t)
      if (lane == lmin && t == tmin) dl[t] = 1e30f;
    ck[r] = bq;
    wk[r] = 1.f / (bd + 1e-8f);
  }
  float ws = wk[0] + wk[1] + wk[2];
  wk[0] /= ws; wk[1] /= ws; wk[2] /= ws;
  const float* f0 = fpr3 + ((size_t)b * N2 + ck[0]) * 256;
  const float* f1 = fpr3 + ((size_t)b * N2 + ck[1]) * 256;
  const float* f2 = fpr3 + ((size_t)b * N2 + ck[2]) * 256;
  half_t* op = out + ((size_t)b * N1 + p) * 416;
  for (int ch = lane; ch < 416; ch += 32) {
    float v = 0.f;
    if (ch < 131) v = ob[(size_t)ch * N1 + p];
    else if (ch < 387) {
      int c2 = ch - 131;
      v = wk[0] * f0[c2] + wk[1] * f1[c2] + wk[2] * f2[c2];
    }
    op[ch] = (half_t)v;
  }
}

// ---------------- 3-NN propagation (coarse 512 -> fine 1024) --------------

__global__ __launch_bounds__(32) void prop1_kernel(
    const float* __restrict__ pts, const float* __restrict__ fpr2 /*[B][512][128]*/,
    half_t* __restrict__ out /*[B][1024][160]*/) {
  int g = blockIdx.x, b = g >> 10, p = g & 1023;
  int lane = (int)threadIdx.x;
  const float* px = pts + (size_t)b * 6 * NPTS;
  float fx = px[p], fy = px[NPTS + p], fz = px[2 * NPTS + p];
  float dl[16];
#pragma unroll
  for (int t = 0; t < 16; ++t) {
    int q = t * 32 + lane;
    float dx = px[q] - fx, dy = px[NPTS + q] - fy, dz = px[2 * NPTS + q] - fz;
    dl[t] = dx * dx + dy * dy + dz * dz;
  }
  float wk[3]; int ck[3];
#pragma unroll
  for (int r = 0; r < 3; ++r) {
    float bd = 1e30f; int bq = 1 << 30;
#pragma unroll
    for (int t = 0; t < 16; ++t) {
      int q = t * 32 + lane;
      if (dl[t] < bd || (dl[t] == bd && q < bq)) { bd = dl[t]; bq = q; }
    }
    wave_argmin(bd, bq);
    int tmin = bq >> 5, lmin = bq & 31;
#pragma unroll
    for (int t = 0; t < 16; ++t)
      if (lane == lmin && t == tmin) dl[t] = 1e30f;
    ck[r] = bq;
    wk[r] = 1.f / (bd + 1e-8f);
  }
  float ws = wk[0] + wk[1] + wk[2];
  wk[0] /= ws; wk[1] /= ws; wk[2] /= ws;
  const float* f0 = fpr2 + ((size_t)b * N1 + ck[0]) * 128;
  const float* f1 = fpr2 + ((size_t)b * N1 + ck[1]) * 128;
  const float* f2 = fpr2 + ((size_t)b * N1 + ck[2]) * 128;
  half_t* op = out + ((size_t)b * NPTS + p) * 160;
  for (int ch = lane; ch < 160; ch += 32) {
    float v = 0.f;
    if (ch < 3) v = px[(size_t)ch * NPTS + p];
    else if (ch < 131) {
      int c2 = ch - 3;
      v = wk[0] * f0[c2] + wk[1] * f1[c2] + wk[2] * f2[c2];
    }
    op[ch] = (half_t)v;
  }
}

// ---------------- host side ----------------------------------------------

static inline void launch_pack(const float* src, half_t* dst, int O, int C,
                               int Opad, int Kpad, hipStream_t s) {
  int n = Opad * Kpad;
  pack_w_kernel<<<(n + 255) / 256, 256, 0, s>>>(src, dst, O, C, Opad, Kpad);
}

static inline void launch_packb(const float* src, float* dst, int O, int Opad,
                                hipStream_t s) {
  pack_b_kernel<<<(Opad + 255) / 256, 256, 0, s>>>(src, dst, O, Opad);
}

static inline void launch_gemm(const half_t* W, const float* bias,
                               const half_t* X, half_t* Yh, float* Yf, int O,
                               int Opad, int Kpad, int P, int relu, int cmajor,
                               hipStream_t s) {
  dim3 grid(P / 64, Opad / 16, BATCH);
  gemm_wmma_kernel<<<grid, 32, 0, s>>>(W, bias, X, Yh, Yf, O, Opad, Kpad, P,
                                       relu, cmajor);
}

extern "C" void kernel_launch(void* const* d_in, const int* in_sizes, int n_in,
                              void* d_out, int out_size, void* d_ws,
                              size_t ws_size, hipStream_t stream) {
  (void)in_sizes; (void)n_in; (void)out_size; (void)ws_size;
  const float* pts = (const float*)d_in[0];

  char* base = (char*)d_ws;
  size_t off = 0;
  auto take = [&](size_t bytes) -> void* {
    void* p = base + off;
    off = (off + bytes + 255) & ~(size_t)255;
    return p;
  };
  // packed f16 weights
  half_t* wr1a = (half_t*)take(sizeof(half_t) * 64 * 32);
  half_t* wr1b = (half_t*)take(sizeof(half_t) * 64 * 64);
  half_t* wr1c = (half_t*)take(sizeof(half_t) * 128 * 64);
  half_t* wr2a = (half_t*)take(sizeof(half_t) * 128 * 160);
  half_t* wr2b = (half_t*)take(sizeof(half_t) * 128 * 128);
  half_t* wr2c = (half_t*)take(sizeof(half_t) * 256 * 128);
  half_t* wr3a = (half_t*)take(sizeof(half_t) * 256 * 288);
  half_t* wr3b = (half_t*)take(sizeof(half_t) * 512 * 256);
  half_t* wr3c = (half_t*)take(sizeof(half_t) * 1024 * 512);
  half_t* wfp3a = (half_t*)take(sizeof(half_t) * 512 * 1312);
  half_t* wfp3b = (half_t*)take(sizeof(half_t) * 256 * 512);
  half_t* wfp2a = (half_t*)take(sizeof(half_t) * 128 * 416);
  half_t* wfp2b = (half_t*)take(sizeof(half_t) * 128 * 128);
  half_t* wfp1a = (half_t*)take(sizeof(half_t) * 128 * 160);
  half_t* wfp1b = (half_t*)take(sizeof(half_t) * 128 * 128);
  half_t* wsega = (half_t*)take(sizeof(half_t) * 128 * 128);
  half_t* wsegb = (half_t*)take(sizeof(half_t) * 32 * 128);
  // padded biases (f32)
  float* br1a = (float*)take(sizeof(float) * 64);
  float* br1b = (float*)take(sizeof(float) * 64);
  float* br1c = (float*)take(sizeof(float) * 128);
  float* br2a = (float*)take(sizeof(float) * 128);
  float* br2b = (float*)take(sizeof(float) * 128);
  float* br2c = (float*)take(sizeof(float) * 256);
  float* br3a = (float*)take(sizeof(float) * 256);
  float* br3b = (float*)take(sizeof(float) * 512);
  float* br3c = (float*)take(sizeof(float) * 1024);
  float* bfp3a = (float*)take(sizeof(float) * 512);
  float* bfp3b = (float*)take(sizeof(float) * 256);
  float* bfp2a = (float*)take(sizeof(float) * 128);
  float* bfp2b = (float*)take(sizeof(float) * 128);
  float* bfp1a = (float*)take(sizeof(float) * 128);
  float* bfp1b = (float*)take(sizeof(float) * 128);
  float* bsega = (float*)take(sizeof(float) * 128);
  float* bsegb = (float*)take(sizeof(float) * 32);
  // intermediates
  int*    idx1  = (int*)take(sizeof(int) * BATCH * N1 * KNN);
  int*    idx2  = (int*)take(sizeof(int) * BATCH * N2 * KNN);
  float*  o1    = (float*)take(sizeof(float) * BATCH * 131 * N1);
  float*  o2f   = (float*)take(sizeof(float) * BATCH * 259 * N2);
  half_t* o2h   = (half_t*)take(sizeof(half_t) * BATCH * N2 * 288);
  half_t* a1    = (half_t*)take(sizeof(half_t) * BATCH * N2 * 256);
  half_t* a2    = (half_t*)take(sizeof(half_t) * BATCH * N2 * 512);
  float*  r3out = (float*)take(sizeof(float) * BATCH * N2 * 1024);
  float*  o3    = (float*)take(sizeof(float) * BATCH * 1024);
  half_t* fp3in = (half_t*)take(sizeof(half_t) * BATCH * N2 * 1312);
  half_t* fp3h  = (half_t*)take(sizeof(half_t) * BATCH * N2 * 512);
  float*  fpr3  = (float*)take(sizeof(float) * BATCH * N2 * 256);
  half_t* fp2in = (half_t*)take(sizeof(half_t) * BATCH * N1 * 416);
  half_t* fp2h  = (half_t*)take(sizeof(half_t) * BATCH * N1 * 128);
  float*  fpr2  = (float*)take(sizeof(float) * BATCH * N1 * 128);
  half_t* fp1in = (half_t*)take(sizeof(half_t) * BATCH * NPTS * 160);
  half_t* fp1h  = (half_t*)take(sizeof(half_t) * BATCH * NPTS * 128);
  half_t* fp1o  = (half_t*)take(sizeof(half_t) * BATCH * NPTS * 128);
  half_t* segh  = (half_t*)take(sizeof(half_t) * BATCH * NPTS * 128);

  // 1) pack weights + biases
  launch_pack((const float*)d_in[1], wr1a, 64, 6, 64, 32, stream);
  launch_pack((const float*)d_in[3], wr1b, 64, 64, 64, 64, stream);
  launch_pack((const float*)d_in[5], wr1c, 128, 64, 128, 64, stream);
  launch_pack((const float*)d_in[7], wr2a, 128, 131, 128, 160, stream);
  launch_pack((const float*)d_in[9], wr2b, 128, 128, 128, 128, stream);
  launch_pack((const float*)d_in[11], wr2c, 256, 128, 256, 128, stream);
  launch_pack((const float*)d_in[13], wr3a, 256, 259, 256, 288, stream);
  launch_pack((const float*)d_in[15], wr3b, 512, 256, 512, 256, stream);
  launch_pack((const float*)d_in[17], wr3c, 1024, 512, 1024, 512, stream);
  launch_pack((const float*)d_in[19], wfp3a, 512, 1283, 512, 1312, stream);
  launch_pack((const float*)d_in[21], wfp3b, 256, 512, 256, 512, stream);
  launch_pack((const float*)d_in[23], wfp2a, 128, 387, 128, 416, stream);
  launch_pack((const float*)d_in[25], wfp2b, 128, 128, 128, 128, stream);
  launch_pack((const float*)d_in[27], wfp1a, 128, 131, 128, 160, stream);
  launch_pack((const float*)d_in[29], wfp1b, 128, 128, 128, 128, stream);
  launch_pack((const float*)d_in[31], wsega, 128, 128, 128, 128, stream);
  launch_pack((const float*)d_in[33], wsegb, 21, 128, 32, 128, stream);
  launch_packb((const float*)d_in[2],  br1a, 64, 64, stream);
  launch_packb((const float*)d_in[4],  br1b, 64, 64, stream);
  launch_packb((const float*)d_in[6],  br1c, 128, 128, stream);
  launch_packb((const float*)d_in[8],  br2a, 128, 128, stream);
  launch_packb((const float*)d_in[10], br2b, 128, 128, stream);
  launch_packb((const float*)d_in[12], br2c, 256, 256, stream);
  launch_packb((const float*)d_in[14], br3a, 256, 256, stream);
  launch_packb((const float*)d_in[16], br3b, 512, 512, stream);
  launch_packb((const float*)d_in[18], br3c, 1024, 1024, stream);
  launch_packb((const float*)d_in[20], bfp3a, 512, 512, stream);
  launch_packb((const float*)d_in[22], bfp3b, 256, 256, stream);
  launch_packb((const float*)d_in[24], bfp2a, 128, 128, stream);
  launch_packb((const float*)d_in[26], bfp2b, 128, 128, stream);
  launch_packb((const float*)d_in[28], bfp1a, 128, 128, stream);
  launch_packb((const float*)d_in[30], bfp1b, 128, 128, stream);
  launch_packb((const float*)d_in[32], bsega, 128, 128, stream);
  launch_packb((const float*)d_in[34], bsegb, 21, 32, stream);

  // 2) set-abstraction stage 1
  knn1_kernel<<<BATCH * N1, 32, 0, stream>>>(pts, idx1, o1);
  stage1_kernel<<<BATCH * N1, 128, 0, stream>>>(pts, idx1, wr1a, br1a, wr1b,
                                                br1b, wr1c, br1c, o1);
  // 3) set-abstraction stage 2
  knn2_kernel<<<BATCH * N2, 32, 0, stream>>>(o1, idx2, o2f, o2h);
  stage2_kernel<<<BATCH * N2, 256, 0, stream>>>(o1, idx2, wr2a, br2a, wr2b,
                                                br2b, wr2c, br2c, o2f, o2h);
  // 4) global stage r3 + max -> o3
  launch_gemm(wr3a, br3a, o2h, a1, nullptr, 256, 256, 288, N2, 1, 0, stream);
  launch_gemm(wr3b, br3b, a1, a2, nullptr, 512, 512, 256, N2, 1, 0, stream);
  launch_gemm(wr3c, br3c, a2, nullptr, r3out, 1024, 1024, 512, N2, 1, 0,
              stream);
  rowmax_kernel<<<(BATCH * 1024 + 255) / 256, 256, 0, stream>>>(r3out, o3, N2,
                                                                1024);
  // 5) fp3
  {
    size_t tot = (size_t)BATCH * N2 * 1312;
    build_fp3in_kernel<<<(unsigned)((tot + 255) / 256), 256, 0, stream>>>(
        o2f, o3, fp3in);
  }
  launch_gemm(wfp3a, bfp3a, fp3in, fp3h, nullptr, 512, 512, 1312, N2, 1, 0,
              stream);
  launch_gemm(wfp3b, bfp3b, fp3h, nullptr, fpr3, 256, 256, 512, N2, 1, 0,
              stream);
  // 6) propagation -> fp2
  prop2_kernel<<<BATCH * N1, 32, 0, stream>>>(o1, fpr3, fp2in);
  launch_gemm(wfp2a, bfp2a, fp2in, fp2h, nullptr, 128, 128, 416, N1, 1, 0,
              stream);
  launch_gemm(wfp2b, bfp2b, fp2h, nullptr, fpr2, 128, 128, 128, N1, 1, 0,
              stream);
  // 7) propagation -> fp1
  prop1_kernel<<<BATCH * NPTS, 32, 0, stream>>>(pts, fpr2, fp1in);
  launch_gemm(wfp1a, bfp1a, fp1in, fp1h, nullptr, 128, 128, 160, NPTS, 1, 0,
              stream);
  launch_gemm(wfp1b, bfp1b, fp1h, fp1o, nullptr, 128, 128, 128, NPTS, 1, 0,
              stream);
  // 8) segmentation head -> d_out (f32, channel-major [B][21][1024])
  launch_gemm(wsega, bsega, fp1o, segh, nullptr, 128, 128, 128, NPTS, 1, 0,
              stream);
  launch_gemm(wsegb, bsegb, segh, nullptr, (float*)d_out, ODIM, 32, 128,
              NPTS, 0, 1, stream);
}
